// VisionMamba_73349451481650
// MI455X (gfx1250) — compile-verified, matrix operand order checked by reading, not run on hardware
//
#include <hip/hip_runtime.h>
#include <hip/hip_bf16.h>

// ---------------------------------------------------------------------------
// VisionMamba forward for MI455X (gfx1250, wave32).
//  * GEMMs: v_wmma_f32_16x16x32_bf16 (bf16 in, f32 accum); weight matrices are
//    zero-padded to N%32==0 at bf16-conversion time so the WMMA inner loop is
//    branch-free (no EXEC juggling, no fragment re-zeroing).
//  * Selective scan: 1 lane = 1 (channel,state); 2 channels / wave32;
//    768 independent waves per direction; per-step chain = 1 FMA.
//  * Conv/softplus/SiLU/RMSNorm/dt_proj stay in f32 VALU.
// ---------------------------------------------------------------------------

#define BB      2
#define TT      16
#define NPATCH_ 196
#define SEQL    (TT * NPATCH_)      // 3136
#define NTOK    (BB * SEQL)         // 6272  (multiple of 64)
#define DM      384
#define DI      768
#define DTR     24
#define NS      16
#define DXP     56                  // dt_rank + 2*n_state
#define DXPP    64                  // DXP padded to 32-multiple (dbl stride)
#define NLAYER  24
#define DCONV   4

typedef __attribute__((ext_vector_type(16))) __bf16 v16bf;
typedef __attribute__((ext_vector_type(8)))  float  v8f;

__device__ __forceinline__ unsigned short f2bf(float f) {
  unsigned int u = __float_as_uint(f);
  u += 0x7FFFu + ((u >> 16) & 1u);          // round-to-nearest-even
  return (unsigned short)(u >> 16);
}

// ---------------------------------------------------------------- im2col ----
__global__ __launch_bounds__(256) void k_im2col(const float* __restrict__ x,
                                                unsigned short* __restrict__ atok) {
  int idx = blockIdx.x * blockDim.x + threadIdx.x;
  if (idx >= NTOK * DI) return;
  int k = idx % DI;                 // c*256 + p*16 + q  (matches patch_w rows)
  int n = idx / DI;
  int c  = k >> 8;
  int p  = (k >> 4) & 15;
  int q  = k & 15;
  int b  = n / SEQL;
  int r  = n % SEQL;
  int t  = r / NPATCH_;
  int pp = r % NPATCH_;
  int hh = pp / 14, ww = pp % 14;
  size_t src = ((((size_t)b * 3 + c) * TT + t) * 224 + (hh * 16 + p)) * 224
               + (ww * 16 + q);
  atok[idx] = f2bf(x[src]);
}

// ------------------------------------------------------------- f32 -> bf16 --
__global__ __launch_bounds__(256) void k_cvt_bf16(const float* __restrict__ in,
                                                  unsigned short* __restrict__ out,
                                                  int n) {
  int i = blockIdx.x * blockDim.x + threadIdx.x;
  if (i < n) out[i] = f2bf(in[i]);
}

// Convert (N x K) weights to bf16, zero-padding rows N..Npad-1.
__global__ __launch_bounds__(256) void k_cvt_bf16_pad(
    const float* __restrict__ in, unsigned short* __restrict__ out,
    int N, int K, int Npad) {
  int i = blockIdx.x * blockDim.x + threadIdx.x;
  if (i >= Npad * K) return;
  int n = i / K;
  out[i] = (n < N) ? f2bf(in[i]) : (unsigned short)0;
}

// -------------------------------------------------------------- WMMA GEMM --
// C[M x Npad](f32, ldc) = A[M x K](bf16) * W[Npad x K](bf16, zero-padded)^T
// M % 64 == 0, Npad % 32 == 0, K % 32 == 0 at every call site.
// Block = 256 threads = 8 waves; wave tile = 32x32 (2x2 WMMA 16x16x32).
// Branch-free inner loop: 8x b128 loads + 4x v_wmma per k-step.
__global__ __launch_bounds__(256) void k_wmma_gemm(
    const unsigned short* __restrict__ A,
    const unsigned short* __restrict__ W,
    float* __restrict__ C, int Npad, int K, int ldc) {
  const int lane = threadIdx.x & 31;
  const int wave = threadIdx.x >> 5;
  const int wm = wave >> 2;                 // 0..1
  const int wn = wave & 3;                  // 0..3
  const int m0 = blockIdx.y * 64 + wm * 32;
  const int n0 = blockIdx.x * 128 + wn * 32;
  if (n0 >= Npad) return;                   // wave-uniform exit for OOB tiles

  const int kh = lane >> 4;                 // which 16-wide K half
  const int lm = lane & 15;

  // v16bf loads are 32B; stride between k-steps = 32 elements = 2 v16bf.
  const v16bf* pA0 = (const v16bf*)(A + (size_t)(m0 + lm) * K + kh * 16);
  const v16bf* pA1 = (const v16bf*)((const unsigned short*)pA0 + (size_t)16 * K);
  const v16bf* pB0 = (const v16bf*)(W + (size_t)(n0 + lm) * K + kh * 16);
  const v16bf* pB1 = (const v16bf*)((const unsigned short*)pB0 + (size_t)16 * K);

  v8f acc00 = {}, acc01 = {}, acc10 = {}, acc11 = {};

  for (int k = 0; k < K; k += 32, pA0 += 2, pA1 += 2, pB0 += 2, pB1 += 2) {
    v16bf a0 = *pA0;
    v16bf a1 = *pA1;
    v16bf b0 = *pB0;
    v16bf b1 = *pB1;
    acc00 = __builtin_amdgcn_wmma_f32_16x16x32_bf16(false, a0, false, b0,
                                                    (short)0, acc00, false, false);
    acc01 = __builtin_amdgcn_wmma_f32_16x16x32_bf16(false, a0, false, b1,
                                                    (short)0, acc01, false, false);
    acc10 = __builtin_amdgcn_wmma_f32_16x16x32_bf16(false, a1, false, b0,
                                                    (short)0, acc10, false, false);
    acc11 = __builtin_amdgcn_wmma_f32_16x16x32_bf16(false, a1, false, b1,
                                                    (short)0, acc11, false, false);
  }

  // C/D layout: VGPR r, lanes 0-15 -> M=r, N=lane; lanes 16-31 -> M=r+8.
  const int rowoff = (lane < 16) ? 0 : 8;
  const int cn = lane & 15;
  const int na = n0 + cn, nb = n0 + 16 + cn;
#pragma unroll
  for (int r = 0; r < 8; ++r) {
    int m = m0 + r + rowoff;
    C[(size_t)m * ldc + na]        = acc00[r];
    C[(size_t)(m + 16) * ldc + na] = acc10[r];
    C[(size_t)m * ldc + nb]        = acc01[r];
    C[(size_t)(m + 16) * ldc + nb] = acc11[r];
  }
}

// --------------------------------------------------- patch-embed epilogue --
__global__ __launch_bounds__(256) void k_tok_finish(float* __restrict__ hidden,
                                                    float* __restrict__ residual,
                                                    const float* __restrict__ pb,
                                                    const float* __restrict__ pos,
                                                    const float* __restrict__ tpos) {
  int idx = blockIdx.x * blockDim.x + threadIdx.x;
  if (idx >= NTOK * DM) return;
  int d = idx % DM;
  int n = idx / DM;
  int r  = n % SEQL;
  int t  = r / NPATCH_;
  int pp = r % NPATCH_;
  hidden[idx] += pb[d] + pos[pp * DM + d] + tpos[t * DM + d];
  residual[idx] = 0.f;
}

// ------------------------------------------ residual add + RMSNorm (bf16) --
__global__ __launch_bounds__(128) void k_resadd_rmsnorm(
    const float* __restrict__ hid, float* __restrict__ res,
    const float* __restrict__ w, unsigned short* __restrict__ hnb) {
  const int n = blockIdx.x;
  const int t = threadIdx.x;
  const size_t base = (size_t)n * DM;
  float v[3];
  float ss = 0.f;
#pragma unroll
  for (int i = 0; i < 3; ++i) {
    int d = t + i * 128;
    float xv = res[base + d] + hid[base + d];
    res[base + d] = xv;
    v[i] = xv;
    ss += xv * xv;
  }
  ss += __shfl_xor(ss, 1);  ss += __shfl_xor(ss, 2);
  ss += __shfl_xor(ss, 4);  ss += __shfl_xor(ss, 8);
  ss += __shfl_xor(ss, 16);
  __shared__ float sm[4];
  if ((t & 31) == 0) sm[t >> 5] = ss;
  __syncthreads();
  float scale = rsqrtf((sm[0] + sm[1] + sm[2] + sm[3]) / (float)DM + 1e-5f);
#pragma unroll
  for (int i = 0; i < 3; ++i) {
    int d = t + i * 128;
    hnb[base + d] = f2bf(v[i] * scale * w[d]);
  }
}

// ---------------------------------------------------- causal conv + SiLU ----
// Produces xc in "frame order": frame index lr; for reverse, source l=L-1-lr.
__global__ __launch_bounds__(256) void k_conv_silu(
    const float* __restrict__ xz, const float* __restrict__ cw,
    const float* __restrict__ cb, float* __restrict__ xc,
    unsigned short* __restrict__ xcb, int reverse) {
  int idx = blockIdx.x * blockDim.x + threadIdx.x;
  if (idx >= NTOK * DI) return;
  int c = idx % DI;
  int n = idx / DI;
  int b = n / SEQL, lr = n % SEQL;
  float acc = cb[c];
#pragma unroll
  for (int k = 0; k < DCONV; ++k) {
    int j = lr - (DCONV - 1) + k;
    if (j >= 0) {
      int ls = reverse ? (SEQL - 1 - j) : j;
      acc += cw[c * DCONV + k] * xz[(size_t)(b * SEQL + ls) * (2 * DI) + c];
    }
  }
  float sig = 1.f / (1.f + __expf(-acc));
  float v = acc * sig;
  xc[idx] = v;
  xcb[idx] = f2bf(v);
}

// ---------------------------------------------------- dt projection (K=24) --
__global__ __launch_bounds__(256) void k_dtproj(const float* __restrict__ dbl,
                                                const float* __restrict__ dw,
                                                const float* __restrict__ db,
                                                float* __restrict__ dt) {
  int idx = blockIdx.x * blockDim.x + threadIdx.x;
  if (idx >= NTOK * DI) return;
  int c = idx % DI;
  int n = idx / DI;
  float a = db[c];
#pragma unroll
  for (int r = 0; r < DTR; ++r) a += dbl[(size_t)n * DXPP + r] * dw[c * DTR + r];
  dt[idx] = (a > 20.f) ? a : log1pf(__expf(a));
}

// ------------------------------------------------------- selective scan -----
// lane = (channel_sub, state): lanes 0-15 -> channel c0 states 0..15,
// lanes 16-31 -> channel c0+1.  768 waves total (B * DI / 2).
__global__ __launch_bounds__(256) void k_scan(
    const float* __restrict__ xc,    // frame order, (B*L) x DI
    const float* __restrict__ dt,    // frame order
    const float* __restrict__ dbl,   // frame order, (B*L) x DXPP
    const float* __restrict__ xz,    // ORIGINAL order, z at +DI
    const float* __restrict__ A_log, // DI x NS
    const float* __restrict__ Ds,    // DI
    float* __restrict__ y,           // ORIGINAL order, (B*L) x DI
    int reverse, int accumulate) {
  const int lane = threadIdx.x & 31;
  const int wid  = (blockIdx.x * blockDim.x + threadIdx.x) >> 5;  // 0..767
  const int s    = lane & 15;
  const int csub = lane >> 4;
  const int b    = wid / (DI / 2);
  const int c    = (wid % (DI / 2)) * 2 + csub;

  const float A  = -__expf(A_log[c * NS + s]);
  const float Dd = Ds[c];
  float h = 0.f;
  const size_t base = (size_t)b * SEQL;

  for (int l = 0; l < SEQL; ++l) {
    const size_t n = base + l;
    if (l + 8 < SEQL) {  // global_prefetch_b8 ahead of the dependent chain
      const size_t np = n + 8;
      __builtin_prefetch(&xc[np * DI + c], 0, 3);
      __builtin_prefetch(&dt[np * DI + c], 0, 3);
      __builtin_prefetch(&dbl[np * DXPP + DTR + s], 0, 3);
    }
    float u  = xc[n * DI + c];
    float d  = dt[n * DI + c];
    float Bv = dbl[n * DXPP + DTR + s];
    float Cv = dbl[n * DXPP + DTR + NS + s];
    h = __expf(d * A) * h + (d * u) * Bv;
    float yp = h * Cv;
    yp += __shfl_xor(yp, 1, 16);
    yp += __shfl_xor(yp, 2, 16);
    yp += __shfl_xor(yp, 4, 16);
    yp += __shfl_xor(yp, 8, 16);
    if (s == 0) {
      const size_t o = base + (reverse ? (SEQL - 1 - l) : l);
      float z   = xz[o * (2 * DI) + DI + c];
      float sig = 1.f / (1.f + __expf(-z));
      float val = (yp + u * Dd) * (z * sig);
      if (accumulate) y[o * DI + c] += val;
      else            y[o * DI + c]  = val;
    }
  }
}

// ------------------------------------------------------ final RMSNorm -------
__global__ __launch_bounds__(128) void k_final_rmsnorm(
    const float* __restrict__ hid, const float* __restrict__ res,
    const float* __restrict__ w, float* __restrict__ out) {
  const int n = blockIdx.x;
  const int t = threadIdx.x;
  const size_t base = (size_t)n * DM;
  float v[3];
  float ss = 0.f;
#pragma unroll
  for (int i = 0; i < 3; ++i) {
    int d = t + i * 128;
    float xv = hid[base + d] + res[base + d];
    v[i] = xv;
    ss += xv * xv;
  }
  ss += __shfl_xor(ss, 1);  ss += __shfl_xor(ss, 2);
  ss += __shfl_xor(ss, 4);  ss += __shfl_xor(ss, 8);
  ss += __shfl_xor(ss, 16);
  __shared__ float sm[4];
  if ((t & 31) == 0) sm[t >> 5] = ss;
  __syncthreads();
  float scale = rsqrtf((sm[0] + sm[1] + sm[2] + sm[3]) / (float)DM + 1e-5f);
#pragma unroll
  for (int i = 0; i < 3; ++i) {
    int d = t + i * 128;
    out[base + d] = v[i] * scale * w[d];
  }
}

// ---------------------------------------------------------------------------
extern "C" void kernel_launch(void* const* d_in, const int* in_sizes, int n_in,
                              void* d_out, int out_size, void* d_ws, size_t ws_size,
                              hipStream_t stream) {
  (void)in_sizes; (void)n_in; (void)out_size; (void)ws_size;
  const float* x        = (const float*)d_in[0];
  const float* patch_w  = (const float*)d_in[2];
  const float* patch_b  = (const float*)d_in[3];
  const float* pos_emb  = (const float*)d_in[4];
  const float* temp_pos = (const float*)d_in[5];
  const float* in_proj  = (const float*)d_in[6];
  const float* conv_w   = (const float*)d_in[7];
  const float* conv_b   = (const float*)d_in[8];
  const float* conv_w_b = (const float*)d_in[9];
  const float* conv_b_b = (const float*)d_in[10];
  const float* xproj    = (const float*)d_in[11];
  const float* xproj_b  = (const float*)d_in[12];
  const float* dtw      = (const float*)d_in[13];
  const float* dtb      = (const float*)d_in[14];
  const float* dtw_b    = (const float*)d_in[15];
  const float* dtb_b    = (const float*)d_in[16];
  const float* A_log    = (const float*)d_in[17];
  const float* A_log_b  = (const float*)d_in[18];
  const float* Ds       = (const float*)d_in[19];
  const float* Ds_b     = (const float*)d_in[20];
  const float* out_proj = (const float*)d_in[21];
  const float* norm_w   = (const float*)d_in[22];
  const float* norm_f_w = (const float*)d_in[23];

  // ---- workspace carve (~145 MB, 256B aligned) ----
  char* wsb = (char*)d_ws;
  size_t off = 0;
  auto carve = [&](size_t bytes) -> void* {
    void* p = wsb + off;
    off = (off + bytes + 255) & ~(size_t)255;
    return p;
  };
  float*          hidden = (float*)carve((size_t)NTOK * DM * 4);
  float*          resid  = (float*)carve((size_t)NTOK * DM * 4);
  unsigned short* hnb    = (unsigned short*)carve((size_t)NTOK * DM * 2);
  float*          xz     = (float*)carve((size_t)NTOK * 2 * DI * 4);
  float*          xc     = (float*)carve((size_t)NTOK * DI * 4);
  unsigned short* xcb    = (unsigned short*)carve((size_t)NTOK * DI * 2);
  float*          dbl    = (float*)carve((size_t)NTOK * DXPP * 4);
  float*          dtv    = (float*)carve((size_t)NTOK * DI * 4);
  float*          yacc   = (float*)carve((size_t)NTOK * DI * 4);
  unsigned short* ybf    = (unsigned short*)carve((size_t)NTOK * DI * 2);
  unsigned short* wbuf   = (unsigned short*)carve((size_t)(2 * DI) * DM * 2);

  const int TPB = 256;
  auto nb = [](long n, int tpb) { return (unsigned)((n + tpb - 1) / tpb); };

  // ---- patch embedding: im2col (bf16) -> WMMA GEMM -> epilogue ----
  k_im2col<<<nb((long)NTOK * DI, TPB), TPB, 0, stream>>>(x, xcb);
  k_cvt_bf16_pad<<<nb(DM * DI, TPB), TPB, 0, stream>>>(patch_w, wbuf, DM, DI, DM);
  {
    dim3 g((DM + 127) / 128, NTOK / 64);
    k_wmma_gemm<<<g, 256, 0, stream>>>(xcb, wbuf, hidden, DM, DI, DM);
  }
  k_tok_finish<<<nb((long)NTOK * DM, TPB), TPB, 0, stream>>>(hidden, resid,
                                                             patch_b, pos_emb,
                                                             temp_pos);

  // ---- 24 Mamba layers ----
  for (int L = 0; L < NLAYER; ++L) {
    k_resadd_rmsnorm<<<NTOK, 128, 0, stream>>>(hidden, resid,
                                               norm_w + (size_t)L * DM, hnb);

    // in_proj: (NTOK x 384) x (1536 x 384)^T -> xz
    k_cvt_bf16_pad<<<nb(2 * DI * DM, TPB), TPB, 0, stream>>>(
        in_proj + (size_t)L * 2 * DI * DM, wbuf, 2 * DI, DM, 2 * DI);
    {
      dim3 g((2 * DI + 127) / 128, NTOK / 64);
      k_wmma_gemm<<<g, 256, 0, stream>>>(hnb, wbuf, xz, 2 * DI, DM, 2 * DI);
    }

    for (int dir = 0; dir < 2; ++dir) {
      const float* cw = (dir ? conv_w_b : conv_w) + (size_t)L * DI * DCONV;
      const float* cb = (dir ? conv_b_b : conv_b) + (size_t)L * DI;
      const float* xp = (dir ? xproj_b : xproj)   + (size_t)L * DXP * DI;
      const float* dW = (dir ? dtw_b : dtw)       + (size_t)L * DI * DTR;
      const float* dB = (dir ? dtb_b : dtb)       + (size_t)L * DI;
      const float* Al = (dir ? A_log_b : A_log)   + (size_t)L * DI * NS;
      const float* Dv = (dir ? Ds_b : Ds)         + (size_t)L * DI;

      k_conv_silu<<<nb((long)NTOK * DI, TPB), TPB, 0, stream>>>(xz, cw, cb, xc,
                                                                xcb, dir);
      // x_proj: (NTOK x 768) x (56 x 768)^T -> dbl (padded 56 -> 64)
      k_cvt_bf16_pad<<<nb(DXPP * DI, TPB), TPB, 0, stream>>>(xp, wbuf, DXP, DI,
                                                             DXPP);
      {
        dim3 g((DXPP + 127) / 128, NTOK / 64);
        k_wmma_gemm<<<g, 256, 0, stream>>>(xcb, wbuf, dbl, DXPP, DI, DXPP);
      }
      k_dtproj<<<nb((long)NTOK * DI, TPB), TPB, 0, stream>>>(dbl, dW, dB, dtv);

      // 768 waves of 32 lanes = 96 blocks of 256
      k_scan<<<(BB * DI / 2) * 32 / 256, 256, 0, stream>>>(
          xc, dtv, dbl, xz, Al, Dv, yacc, dir, dir);
    }

    // out_proj: (NTOK x 768) x (384 x 768)^T -> hidden
    k_cvt_bf16<<<nb((long)NTOK * DI, TPB), TPB, 0, stream>>>(yacc, ybf,
                                                             NTOK * DI);
    k_cvt_bf16_pad<<<nb(DM * DI, TPB), TPB, 0, stream>>>(
        out_proj + (size_t)L * DM * DI, wbuf, DM, DI, DM);
    {
      dim3 g((DM + 127) / 128, NTOK / 64);
      k_wmma_gemm<<<g, 256, 0, stream>>>(ybf, wbuf, hidden, DM, DI, DM);
    }
  }

  k_final_rmsnorm<<<NTOK, 128, 0, stream>>>(hidden, resid, norm_f_w,
                                            (float*)d_out);
}
